// ConvNet_transformer_80479097193190
// MI455X (gfx1250) — compile-verified, matrix operand order checked by reading, not run on hardware
//
#include <hip/hip_runtime.h>
#include <math.h>

// ---------------------------------------------------------------------------
// ConvNet + Transformer + GAT + edge regression for MI455X (gfx1250, wave32).
// All dense layers run through a WMMA f16 GEMM (f32 accumulate), with
// double-buffered LDS staging and 2 independent accumulator tiles per wave.
// ---------------------------------------------------------------------------

typedef _Float16 h16v __attribute__((ext_vector_type(16)));
typedef _Float16 h8v  __attribute__((ext_vector_type(8)));
typedef float    f8v  __attribute__((ext_vector_type(8)));

#define EPS_BN 1e-5f
#define EPS_LN 1e-5f

static __device__ __forceinline__ unsigned fmax_map(float x) {
  unsigned u = __float_as_uint(x);
  return (u & 0x80000000u) ? ~u : (u | 0x80000000u);
}
static __device__ __forceinline__ float fmax_unmap(unsigned u) {
  unsigned b = (u & 0x80000000u) ? (u & 0x7fffffffu) : ~u;
  return __uint_as_float(b);
}

// ============================ WMMA GEMM ====================================
// C[m,n] = act( A[m,:K] . W[n,:K] + bias[n] + res[m,n] )
// A: [M,K] row-major (lda), W: [N,K] row-major.
// Requirements: M%32==0, N%128==0, K%32==0.
// Block = 256 threads = 8 waves; macro tile 32(M) x 128(N).
// Waves arranged 2(M) x 4(N); each wave computes a 16x32 tile -> two
// independent 16x16 WMMA accumulator chains (hides XDL latency, reuses the
// A fragment). LDS double-buffered: one workgroup barrier per K-step.
#define LDS_K   40   // halves per row; 80B rows keep 16B alignment, conflict-free

__global__ __launch_bounds__(256) void k_gemm_wmma(
    const float* __restrict__ A, int lda,
    const float* __restrict__ W,
    const float* __restrict__ bias,
    const float* __restrict__ res, int ldres,
    float* __restrict__ C, int ldc,
    int K, int act)
{
  __shared__ __align__(16) _Float16 As[2][32 * LDS_K];
  __shared__ __align__(16) _Float16 Ws[2][128 * LDS_K];

  const int tid    = threadIdx.x;
  const int lane   = tid & 31;
  const int wave   = tid >> 5;
  const int waveM  = wave >> 2;       // 0..1
  const int waveN  = wave & 3;        // 0..3
  const int laneLo = lane & 15;
  const int laneHi = lane >> 4;

  const int mBlock = blockIdx.y * 32;
  const int nBlock = blockIdx.x * 128;

  // staging assignments
  const int arow   = tid >> 3;          // A: 32 rows, 8 threads/row
  const int acol4  = (tid & 7) << 2;    //    4 floats each
  const int wrow   = tid >> 1;          // W: 128 rows, 2 threads/row
  const int wcol16 = (tid & 1) << 4;    //    16 floats each

  const float* Aptr = A + (size_t)(mBlock + arow) * lda + acol4;
  const float* Wptr = W + (size_t)(nBlock + wrow) * K + wcol16;

  f8v acc0 = {};
  f8v acc1 = {};

  // fragment source pointers within a buffer
  const int aoff  = (waveM * 16 + laneLo) * LDS_K + laneHi * 8;
  const int boff0 = (waveN * 32 + laneLo) * LDS_K + laneHi * 8;
  const int boff1 = boff0 + 16 * LDS_K;

#define STAGE_TILE(b)                                                        \
  do {                                                                       \
    _Float16* ad = &As[b][arow * LDS_K + acol4];                             \
    ad[0] = (_Float16)av.x; ad[1] = (_Float16)av.y;                          \
    ad[2] = (_Float16)av.z; ad[3] = (_Float16)av.w;                          \
    _Float16* wd = &Ws[b][wrow * LDS_K + wcol16];                            \
    wd[0]  = (_Float16)w0.x; wd[1]  = (_Float16)w0.y;                        \
    wd[2]  = (_Float16)w0.z; wd[3]  = (_Float16)w0.w;                        \
    wd[4]  = (_Float16)w1.x; wd[5]  = (_Float16)w1.y;                        \
    wd[6]  = (_Float16)w1.z; wd[7]  = (_Float16)w1.w;                        \
    wd[8]  = (_Float16)w2.x; wd[9]  = (_Float16)w2.y;                        \
    wd[10] = (_Float16)w2.z; wd[11] = (_Float16)w2.w;                        \
    wd[12] = (_Float16)w3.x; wd[13] = (_Float16)w3.y;                        \
    wd[14] = (_Float16)w3.z; wd[15] = (_Float16)w3.w;                        \
  } while (0)

#define LOAD_TILE()                                                          \
  float4 av = *(const float4*)Aptr;                                          \
  float4 w0 = *(const float4*)(Wptr + 0);                                    \
  float4 w1 = *(const float4*)(Wptr + 4);                                    \
  float4 w2 = *(const float4*)(Wptr + 8);                                    \
  float4 w3 = *(const float4*)(Wptr + 12);                                   \
  Aptr += 32; Wptr += 32;                                                    \
  __builtin_prefetch((const void*)Aptr, 0, 3);                               \
  __builtin_prefetch((const void*)Wptr, 0, 3)

#define COMPUTE(b)                                                           \
  do {                                                                       \
    const _Float16* ar = &As[b][aoff];                                       \
    h8v alo = *(const h8v*)(ar);                                             \
    h8v ahi = *(const h8v*)(ar + 16);                                        \
    h16v af = __builtin_shufflevector(alo, ahi,                              \
        0,1,2,3,4,5,6,7,8,9,10,11,12,13,14,15);                              \
    const _Float16* br0 = &Ws[b][boff0];                                     \
    h8v b0lo = *(const h8v*)(br0);                                           \
    h8v b0hi = *(const h8v*)(br0 + 16);                                      \
    h16v bf0 = __builtin_shufflevector(b0lo, b0hi,                           \
        0,1,2,3,4,5,6,7,8,9,10,11,12,13,14,15);                              \
    const _Float16* br1 = &Ws[b][boff1];                                     \
    h8v b1lo = *(const h8v*)(br1);                                           \
    h8v b1hi = *(const h8v*)(br1 + 16);                                      \
    h16v bf1 = __builtin_shufflevector(b1lo, b1hi,                           \
        0,1,2,3,4,5,6,7,8,9,10,11,12,13,14,15);                              \
    acc0 = __builtin_amdgcn_wmma_f32_16x16x32_f16(                           \
        false, af, false, bf0, (short)0, acc0, false, false);                \
    acc1 = __builtin_amdgcn_wmma_f32_16x16x32_f16(                           \
        false, af, false, bf1, (short)0, acc1, false, false);                \
  } while (0)

  // preload tile 0 into buffer 0
  {
    LOAD_TILE();
    STAGE_TILE(0);
  }
  __syncthreads();

  int buf = 0;
  for (int k0 = 32; k0 < K; k0 += 32) {
    LOAD_TILE();          // global loads for tile k0 overlap with WMMAs below
    COMPUTE(buf);
    STAGE_TILE(buf ^ 1);  // disjoint buffer: no barrier needed before stores
    __syncthreads();      // single barrier per K-step
    buf ^= 1;
  }
  COMPUTE(buf);

#undef STAGE_TILE
#undef LOAD_TILE
#undef COMPUTE

  // epilogue: lane(0..15)=N within tile; vgpr e -> M = e + 8*laneHi
  const int n0 = nBlock + waveN * 32 + laneLo;
  const int n1 = n0 + 16;
  const float bv0 = bias ? bias[n0] : 0.0f;
  const float bv1 = bias ? bias[n1] : 0.0f;
#pragma unroll
  for (int e = 0; e < 8; ++e) {
    const int m = mBlock + waveM * 16 + e + laneHi * 8;
    float v0 = acc0[e] + bv0;
    float v1 = acc1[e] + bv1;
    if (res) {
      v0 += res[(size_t)m * ldres + n0];
      v1 += res[(size_t)m * ldres + n1];
    }
    if (act) { v0 = fmaxf(v0, 0.0f); v1 = fmaxf(v1, 0.0f); }
    C[(size_t)m * ldc + n0] = v0;
    C[(size_t)m * ldc + n1] = v1;
  }
}

// ============================ CNN kernels ==================================

__global__ void k_conv0_relu(const float* __restrict__ in,
                             const float* __restrict__ w,
                             const float* __restrict__ b,
                             float* __restrict__ out,
                             int H, int Wd, int total)
{
  int idx = blockIdx.x * blockDim.x + threadIdx.x;
  if (idx >= total) return;
  int x = idx % Wd;
  int y = (idx / Wd) % H;
  int co = (idx / (Wd * H)) % 16;
  int n = idx / (Wd * H * 16);
  const float* wp = w + co * 9;
  const float* ip = in + (size_t)n * H * Wd;
  float acc = b[co];
  for (int ky = 0; ky < 3; ++ky) {
    int yy = y + ky - 1;
    if (yy < 0 || yy >= H) continue;
    for (int kx = 0; kx < 3; ++kx) {
      int xx = x + kx - 1;
      if (xx < 0 || xx >= Wd) continue;
      acc += ip[yy * Wd + xx] * wp[ky * 3 + kx];
    }
  }
  out[idx] = fmaxf(acc, 0.0f);
}

// fused conv3x3(pad1) -> BN(eval) -> relu -> maxpool(k2,s2,p1)
__global__ void k_conv_bn_relu_pool(const float* __restrict__ in,
                                    const float* __restrict__ w,
                                    const float* __restrict__ bias,
                                    const float* __restrict__ gamma,
                                    const float* __restrict__ beta,
                                    float* __restrict__ out,
                                    int CIN, int COUT, int HIN, int WIN,
                                    int HOUT, int WOUT, int total)
{
  int idx = blockIdx.x * blockDim.x + threadIdx.x;
  if (idx >= total) return;
  int px = idx % WOUT;
  int t = idx / WOUT;
  int py = t % HOUT; t /= HOUT;
  int co = t % COUT;
  int n = t / COUT;
  const float* wb = w + co * CIN * 9;
  const float scale = gamma[co] * rsqrtf(1.0f + EPS_BN);
  const float shift = beta[co];
  const float bb = bias[co];
  float best = 0.0f;  // relu(.) >= 0 and window non-empty
  for (int wy = 0; wy < 2; ++wy) {
    int y = py * 2 - 1 + wy;
    if (y < 0 || y >= HIN) continue;
    for (int wx = 0; wx < 2; ++wx) {
      int x = px * 2 - 1 + wx;
      if (x < 0 || x >= WIN) continue;
      float acc = 0.0f;
      for (int ci = 0; ci < CIN; ++ci) {
        const float* ip = in + (size_t)(n * CIN + ci) * HIN * WIN;
        const float* wp = wb + ci * 9;
        for (int ky = 0; ky < 3; ++ky) {
          int yy = y + ky - 1;
          if (yy < 0 || yy >= HIN) continue;
          for (int kx = 0; kx < 3; ++kx) {
            int xx = x + kx - 1;
            if (xx < 0 || xx >= WIN) continue;
            acc += ip[yy * WIN + xx] * wp[ky * 3 + kx];
          }
        }
      }
      float v = fmaxf((acc + bb) * scale + shift, 0.0f);
      best = fmaxf(best, v);
    }
  }
  out[idx] = best;
}

__global__ void k_chanmean(const float* __restrict__ in, float* __restrict__ out, int HW)
{
  int nc = blockIdx.x;
  const float* p = in + (size_t)nc * HW;
  __shared__ float red[256];
  float s = 0.0f;
  for (int i = threadIdx.x; i < HW; i += 256) s += p[i];
  red[threadIdx.x] = s;
  __syncthreads();
  for (int off = 128; off > 0; off >>= 1) {
    if (threadIdx.x < off) red[threadIdx.x] += red[threadIdx.x + off];
    __syncthreads();
  }
  if (threadIdx.x == 0) out[nc] = red[0] / (float)HW;
}

// SE MLP: scale[n,c] = sigmoid(W2 relu(W1 mean + b1) + b2)
__global__ void k_se(const float* __restrict__ mean,
                     const float* __restrict__ w1, const float* __restrict__ b1,
                     const float* __restrict__ w2, const float* __restrict__ b2,
                     float* __restrict__ scale, int C, int CH)
{
  int n = blockIdx.x;
  int t = threadIdx.x;
  __shared__ float hid[4];
  __shared__ float mloc[32];
  if (t < C) mloc[t] = mean[n * C + t];
  __syncthreads();
  if (t < CH) {
    float a = b1[t];
    for (int c = 0; c < C; ++c) a += mloc[c] * w1[t * C + c];
    hid[t] = fmaxf(a, 0.0f);
  }
  __syncthreads();
  if (t < C) {
    float a = b2[t];
    for (int h = 0; h < CH; ++h) a += hid[h] * w2[t * CH + h];
    scale[n * C + t] = 1.0f / (1.0f + expf(-a));
  }
}

__global__ void k_scale_inplace(float* __restrict__ a, const float* __restrict__ s,
                                int C, int HW, int total)
{
  int idx = blockIdx.x * blockDim.x + threadIdx.x;
  if (idx >= total) return;
  int c = (idx / HW) % C;
  int n = idx / (HW * C);
  a[idx] *= s[n * C + c];
}

// ============================ misc elementwise =============================

__global__ void k_layernorm256(float* __restrict__ io,
                               const float* __restrict__ g,
                               const float* __restrict__ b)
{
  int row = blockIdx.x;
  int t = threadIdx.x;  // blockDim == 256 == D
  float* p = io + (size_t)row * 256;
  __shared__ float red[256];
  __shared__ float s_mean, s_rstd;
  float v = p[t];
  red[t] = v;
  __syncthreads();
  for (int off = 128; off > 0; off >>= 1) {
    if (t < off) red[t] += red[t + off];
    __syncthreads();
  }
  if (t == 0) s_mean = red[0] * (1.0f / 256.0f);
  __syncthreads();
  float d = v - s_mean;
  red[t] = d * d;
  __syncthreads();
  for (int off = 128; off > 0; off >>= 1) {
    if (t < off) red[t] += red[t + off];
    __syncthreads();
  }
  if (t == 0) s_rstd = rsqrtf(red[0] * (1.0f / 256.0f) + EPS_LN);
  __syncthreads();
  p[t] = d * s_rstd * g[t] + b[t];
}

__global__ void k_add_act(float* __restrict__ out, const float* __restrict__ a,
                          const float* __restrict__ b, int total, int act)
{
  int idx = blockIdx.x * blockDim.x + threadIdx.x;
  if (idx >= total) return;
  float v = a[idx] + b[idx];
  if (act) v = fmaxf(v, 0.0f);
  out[idx] = v;
}

__global__ void k_rowbias_act(float* __restrict__ out, const float* __restrict__ in,
                              const float* __restrict__ bias, int total, int cols, int act)
{
  int idx = blockIdx.x * blockDim.x + threadIdx.x;
  if (idx >= total) return;
  float v = in[idx] + bias[idx % cols];
  if (act) v = fmaxf(v, 0.0f);
  out[idx] = v;
}

__global__ void k_fill_f(float* __restrict__ p, float v, int total)
{
  int idx = blockIdx.x * blockDim.x + threadIdx.x;
  if (idx < total) p[idx] = v;
}
__global__ void k_fill_u(unsigned* __restrict__ p, unsigned v, int total)
{
  int idx = blockIdx.x * blockDim.x + threadIdx.x;
  if (idx < total) p[idx] = v;
}

__global__ void k_transpose(const float* __restrict__ in, float* __restrict__ out,
                            int R, int Ccols)
{
  int idx = blockIdx.x * blockDim.x + threadIdx.x;
  if (idx >= R * Ccols) return;
  int c = idx % Ccols;
  int r = idx / Ccols;
  out[c * R + r] = in[idx];
}

// ============================ GAT kernels ==================================

static __device__ __forceinline__ void edge_sd(const int* __restrict__ ei, int e,
                                               int& s, int& d)
{
  if (e < 8192) { s = ei[e]; d = ei[8192 + e]; }
  else          { s = d = e - 8192; }
}

__global__ void k_gat_coef(const float* __restrict__ h,
                           const float* __restrict__ a_src,
                           const float* __restrict__ a_dst,
                           float* __restrict__ asrc, float* __restrict__ adst)
{
  int idx = blockIdx.x * blockDim.x + threadIdx.x;  // node*4 + head
  if (idx >= 256 * 4) return;
  int node = idx >> 2;
  int hh = idx & 3;
  const float* hp = h + (size_t)node * 256 + hh * 64;
  const float* asp = a_src + hh * 64;
  const float* adp = a_dst + hh * 64;
  float s0 = 0.0f, s1 = 0.0f;
  for (int d = 0; d < 64; ++d) {
    float hv = hp[d];
    s0 += hv * asp[d];
    s1 += hv * adp[d];
  }
  asrc[idx] = s0;
  adst[idx] = s1;
}

__global__ void k_edge_max(const int* __restrict__ ei,
                           const float* __restrict__ asrc,
                           const float* __restrict__ adst,
                           unsigned* __restrict__ m, int total)
{
  int idx = blockIdx.x * blockDim.x + threadIdx.x;  // e*4 + head
  if (idx >= total) return;
  int e = idx >> 2, hh = idx & 3;
  int s, d;
  edge_sd(ei, e, s, d);
  float v = asrc[s * 4 + hh] + adst[d * 4 + hh];
  v = (v > 0.0f) ? v : 0.2f * v;  // leaky_relu 0.2
  atomicMax(&m[d * 4 + hh], fmax_map(v));
}

__global__ void k_edge_exp(const int* __restrict__ ei,
                           const float* __restrict__ asrc,
                           const float* __restrict__ adst,
                           const unsigned* __restrict__ m,
                           float* __restrict__ ee, float* __restrict__ denom, int total)
{
  int idx = blockIdx.x * blockDim.x + threadIdx.x;
  if (idx >= total) return;
  int e = idx >> 2, hh = idx & 3;
  int s, d;
  edge_sd(ei, e, s, d);
  float v = asrc[s * 4 + hh] + adst[d * 4 + hh];
  v = (v > 0.0f) ? v : 0.2f * v;
  float mv = fmax_unmap(m[d * 4 + hh]);
  float x = expf(v - mv);
  ee[idx] = x;
  atomicAdd(&denom[d * 4 + hh], x);
}

__global__ void k_edge_agg(const int* __restrict__ ei,
                           const float* __restrict__ ee,
                           const float* __restrict__ denom,
                           const float* __restrict__ h,
                           float* __restrict__ agg, int total)
{
  int idx = blockIdx.x * blockDim.x + threadIdx.x;  // e*256 + j
  if (idx >= total) return;
  int j = idx & 255;
  int e = idx >> 8;
  int hh = j >> 6;
  int s, d;
  edge_sd(ei, e, s, d);
  float alpha = ee[e * 4 + hh] / (denom[d * 4 + hh] + 1e-16f);
  atomicAdd(&agg[d * 256 + j], alpha * h[s * 256 + j]);
}

// ============================ edge features / regression ===================

__global__ void k_build_xe(const int* __restrict__ ei,
                           const float* __restrict__ xt,
                           const float* __restrict__ xr,
                           float* __restrict__ xe, int total)
{
  int idx = blockIdx.x * blockDim.x + threadIdx.x;  // e*1024 + j
  if (idx >= total) return;
  int j = idx & 1023;
  int e = idx >> 10;
  int node = (j < 512) ? ei[e] : ei[8192 + e];
  int jj = j & 511;
  float v = (jj < 256) ? xt[node * 256 + jj] : xr[node * 256 + (jj - 256)];
  xe[idx] = v;
}

__global__ void k_reg_final(const float* __restrict__ hid,
                            const float* __restrict__ w4,
                            const float* __restrict__ b4,
                            float* __restrict__ y)
{
  int wv = (blockIdx.x * blockDim.x + threadIdx.x) >> 5;  // wave per edge
  int lane = threadIdx.x & 31;
  if (wv >= 8192) return;
  const float* hp = hid + (size_t)wv * 512;
  float s = 0.0f;
  for (int k = lane; k < 512; k += 32) s += hp[k] * w4[k];
  for (int off = 16; off > 0; off >>= 1) s += __shfl_xor(s, off, 32);
  if (lane == 0) y[wv] = s + b4[0];
}

// ============================ host driver ==================================

extern "C" void kernel_launch(void* const* d_in, const int* in_sizes, int n_in,
                              void* d_out, int out_size, void* d_ws, size_t ws_size,
                              hipStream_t stream)
{
  (void)in_sizes; (void)n_in; (void)out_size; (void)ws_size;

  // ---- input mapping: JAX pytree flatten, dict keys sorted recursively ----
  const int*   edge_index = (const int*)d_in[0];
  const float* cnn_b[4]    = {(const float*)d_in[1],  (const float*)d_in[5],
                              (const float*)d_in[9],  (const float*)d_in[13]};
  const float* cnn_beta[4] = {(const float*)d_in[2],  (const float*)d_in[6],
                              (const float*)d_in[10], (const float*)d_in[14]};
  const float* cnn_g[4]    = {(const float*)d_in[3],  (const float*)d_in[7],
                              (const float*)d_in[11], (const float*)d_in[15]};
  const float* cnn_w[4]    = {(const float*)d_in[4],  (const float*)d_in[8],
                              (const float*)d_in[12], (const float*)d_in[16]};
  const float* conv0_b = (const float*)d_in[17];
  const float* conv0_w = (const float*)d_in[18];
  const float* fc_b    = (const float*)d_in[19];
  const float* fc_w    = (const float*)d_in[20];
  const float* gat1_ad = (const float*)d_in[21];
  const float* gat1_as = (const float*)d_in[22];
  const float* gat1_b  = (const float*)d_in[23];
  const float* gat1_w  = (const float*)d_in[24];
  const float* gatN_ad = (const float*)d_in[25];
  const float* gatN_as = (const float*)d_in[26];
  const float* gatN_b  = (const float*)d_in[27];
  const float* gatN_w  = (const float*)d_in[28];
  const float* reg_b1  = (const float*)d_in[29];
  const float* reg_b4  = (const float*)d_in[30];
  const float* reg_w1  = (const float*)d_in[31];
  const float* reg_w4  = (const float*)d_in[32];
  const float* se1_b1  = (const float*)d_in[33];
  const float* se1_b2  = (const float*)d_in[34];
  const float* se1_w1  = (const float*)d_in[35];
  const float* se1_w2  = (const float*)d_in[36];
  const float* se2_b1  = (const float*)d_in[37];
  const float* se2_b2  = (const float*)d_in[38];
  const float* se2_w1  = (const float*)d_in[39];
  const float* se2_w2  = (const float*)d_in[40];
  // tr[l] leaves sorted: b1,b2,bo,bqkv,ln1_b,ln1_g,ln2_b,ln2_g,w1,w2,wo,wqkv
  auto TR = [&](int l, int i) { return (const float*)d_in[41 + l * 12 + i]; };
  const float* x = (const float*)d_in[77];

  // ---- workspace layout (floats), ~113 MB total ----
  float* wsp = (float*)d_ws;
  size_t off = 0;
  auto alloc = [&](size_t n) {
    float* p = wsp + off;
    off += (n + 255) & ~(size_t)255;
    return p;
  };
  // CNN chunk buffers (chunk = 32 images: largest buffer 64.6MB -> L2 resident)
  const int CH_IMG = 32;
  float* c0 = alloc((size_t)CH_IMG * 16 * 322 * 98);
  float* p1 = alloc((size_t)CH_IMG * 16 * 162 * 50);
  float* p2 = alloc((size_t)CH_IMG * 32 * 82 * 26);
  float* p3 = alloc((size_t)CH_IMG * 32 * 42 * 14);
  float* p4 = alloc((size_t)CH_IMG * 32 * 22 * 8);
  float* semean  = alloc(1024);
  float* sescale = alloc(1024);
  float* feat = alloc(256 * 256);
  float* bufV = alloc(256 * 256);
  float* bufX = alloc(256 * 256);
  float* bufF = alloc(256 * 512);
  float* bufY = alloc(256 * 256);
  float* xt   = alloc(256 * 256);
  float* xr   = alloc(256 * 256);
  float* wT   = alloc(256 * 256);
  float* hbuf = alloc(256 * 256);
  float* hin2 = alloc(256 * 256);
  float* asrc = alloc(1024);
  float* adst = alloc(1024);
  unsigned* mbuf = (unsigned*)alloc(1024);
  float* denom = alloc(1024);
  float* eebuf = alloc((size_t)8448 * 4);
  float* agg  = alloc(256 * 256);
  float* g1   = alloc(256 * 256);
  float* rhid = alloc((size_t)8192 * 512);

  auto blocks = [](long n) { return dim3((unsigned)((n + 255) / 256)); };
  auto gemm = [&](const float* A, int lda, const float* Wm, const float* bias,
                  const float* res, int ldres, float* C, int ldc,
                  int M, int N, int K, int act) {
    dim3 grid(N / 128, M / 32);
    k_gemm_wmma<<<grid, 256, 0, stream>>>(A, lda, Wm, bias, res, ldres, C, ldc, K, act);
  };

  // ======================= SE_CNN (8 chunks of 32 images) =================
  for (int ch = 0; ch < 8; ++ch) {
    const float* xin = x + (size_t)ch * CH_IMG * 322 * 98;
    int t0 = CH_IMG * 16 * 322 * 98;
    k_conv0_relu<<<blocks(t0), 256, 0, stream>>>(xin, conv0_w, conv0_b, c0, 322, 98, t0);
    k_chanmean<<<CH_IMG * 16, 256, 0, stream>>>(c0, semean, 322 * 98);
    k_se<<<CH_IMG, 32, 0, stream>>>(semean, se1_w1, se1_b1, se1_w2, se1_b2, sescale, 16, 1);
    k_scale_inplace<<<blocks(t0), 256, 0, stream>>>(c0, sescale, 16, 322 * 98, t0);

    int t1 = CH_IMG * 16 * 162 * 50;
    k_conv_bn_relu_pool<<<blocks(t1), 256, 0, stream>>>(
        c0, cnn_w[0], cnn_b[0], cnn_g[0], cnn_beta[0], p1, 16, 16, 322, 98, 162, 50, t1);
    int t2 = CH_IMG * 32 * 82 * 26;
    k_conv_bn_relu_pool<<<blocks(t2), 256, 0, stream>>>(
        p1, cnn_w[1], cnn_b[1], cnn_g[1], cnn_beta[1], p2, 16, 32, 162, 50, 82, 26, t2);

    k_chanmean<<<CH_IMG * 32, 256, 0, stream>>>(p2, semean, 82 * 26);
    k_se<<<CH_IMG, 32, 0, stream>>>(semean, se2_w1, se2_b1, se2_w2, se2_b2, sescale, 32, 2);
    k_scale_inplace<<<blocks(t2), 256, 0, stream>>>(p2, sescale, 32, 82 * 26, t2);

    int t3 = CH_IMG * 32 * 42 * 14;
    k_conv_bn_relu_pool<<<blocks(t3), 256, 0, stream>>>(
        p2, cnn_w[2], cnn_b[2], cnn_g[2], cnn_beta[2], p3, 32, 32, 82, 26, 42, 14, t3);
    int t4 = CH_IMG * 32 * 22 * 8;
    k_conv_bn_relu_pool<<<blocks(t4), 256, 0, stream>>>(
        p3, cnn_w[3], cnn_b[3], cnn_g[3], cnn_beta[3], p4, 32, 32, 42, 14, 22, 8, t4);

    // fc: [32,5632] @ [256,5632]^T  -> feat rows
    gemm(p4, 5632, fc_w, fc_b, nullptr, 0,
         feat + (size_t)ch * CH_IMG * 256, 256, CH_IMG, 256, 5632, 0);
  }

  // ======================= Transformer (S==1 -> MHA(x)=Wo(Wv x)) ==========
  const float* cur = feat;
  for (int l = 0; l < 3; ++l) {
    const float* b1q = TR(l, 0), *b2q = TR(l, 1), *bo = TR(l, 2), *bqkv = TR(l, 3);
    const float* ln1b = TR(l, 4), *ln1g = TR(l, 5), *ln2b = TR(l, 6), *ln2g = TR(l, 7);
    const float* w1q = TR(l, 8), *w2q = TR(l, 9), *wo = TR(l, 10), *wqkv = TR(l, 11);
    // v = x @ Wv^T + bv   (rows 512..767 of wqkv)
    gemm(cur, 256, wqkv + 512 * 256, bqkv + 512, nullptr, 0, bufV, 256, 256, 256, 256, 0);
    // x + MHA(x) = x + v @ Wo^T + bo
    gemm(bufV, 256, wo, bo, cur, 256, bufX, 256, 256, 256, 256, 0);
    k_layernorm256<<<256, 256, 0, stream>>>(bufX, ln1g, ln1b);
    // FFN
    gemm(bufX, 256, w1q, b1q, nullptr, 0, bufF, 512, 256, 512, 256, 1);
    gemm(bufF, 512, w2q, b2q, bufX, 256, bufY, 256, 256, 256, 512, 0);
    k_layernorm256<<<256, 256, 0, stream>>>(bufY, ln2g, ln2b);
    cur = bufY;
  }
  k_add_act<<<blocks(65536), 256, 0, stream>>>(xt, cur, feat, 65536, 1);  // x_t

  // ======================= MultiGAT (2 layers) ============================
  const int ETOT = 8192 + 256;
  auto gat_layer = [&](const float* xin, const float* wparam,
                       const float* a_s, const float* a_d) {
    k_transpose<<<blocks(65536), 256, 0, stream>>>(wparam, wT, 256, 256);
    gemm(xin, 256, wT, nullptr, nullptr, 0, hbuf, 256, 256, 256, 256, 0);
    k_gat_coef<<<blocks(1024), 256, 0, stream>>>(hbuf, a_s, a_d, asrc, adst);
    k_fill_u<<<blocks(1024), 256, 0, stream>>>(mbuf, 0u, 1024);
    k_fill_f<<<blocks(1024), 256, 0, stream>>>(denom, 0.0f, 1024);
    k_fill_f<<<blocks(65536), 256, 0, stream>>>(agg, 0.0f, 65536);
    k_edge_max<<<blocks(ETOT * 4), 256, 0, stream>>>(edge_index, asrc, adst, mbuf, ETOT * 4);
    k_edge_exp<<<blocks(ETOT * 4), 256, 0, stream>>>(edge_index, asrc, adst, mbuf,
                                                     eebuf, denom, ETOT * 4);
    k_edge_agg<<<blocks((long)ETOT * 256), 256, 0, stream>>>(edge_index, eebuf, denom,
                                                             hbuf, agg, ETOT * 256);
  };
  gat_layer(feat, gat1_w, gat1_as, gat1_ad);
  k_rowbias_act<<<blocks(65536), 256, 0, stream>>>(g1, agg, gat1_b, 65536, 256, 1);
  k_add_act<<<blocks(65536), 256, 0, stream>>>(hin2, g1, feat, 65536, 0);
  gat_layer(hin2, gatN_w, gatN_as, gatN_ad);
  k_rowbias_act<<<blocks(65536), 256, 0, stream>>>(g1, agg, gatN_b, 65536, 256, 0);
  k_add_act<<<blocks(65536), 256, 0, stream>>>(xr, g1, feat, 65536, 1);  // x_r

  // ======================= edge features + regression =====================
  float* ypre = (float*)d_out;
  float* xe = ypre + 8192;  // xe written straight into d_out
  k_build_xe<<<blocks((long)8192 * 1024), 256, 0, stream>>>(edge_index, xt, xr, xe,
                                                            8192 * 1024);
  gemm(xe, 1024, reg_w1, reg_b1, nullptr, 0, rhid, 512, 8192, 512, 1024, 1);
  k_reg_final<<<1024, 256, 0, stream>>>(rhid, reg_w4, reg_b4, ypre);
}
